// SelfAtt_34316788695626
// MI455X (gfx1250) — matrix operand order repaired and checked by reading, not compiled, not run on hardware
//
#include <hip/hip_runtime.h>
#include <hip/hip_bf16.h>

// ---------------------------------------------------------------------------
// Self-attention for MI455X (gfx1250): bf16 WMMA everywhere, flash-attention
// fusion, V pre-transposed, K/V tiles streamed into LDS by the Tensor Data
// Mover (double-buffered, s_wait_tensorcnt), Wo pre-converted to bf16.
// ---------------------------------------------------------------------------

#define Bx 8
#define Lx 1024
#define Ex 1024
#define Hx 16
#define Dx 64

typedef __attribute__((ext_vector_type(16))) __bf16 v16bf;
typedef __attribute__((ext_vector_type(8)))  float  v8f;
typedef __attribute__((ext_vector_type(4)))  unsigned int u32x4;
typedef __attribute__((ext_vector_type(4)))  int i32x4;
typedef __attribute__((ext_vector_type(8)))  int i32x8;

union Frag {
    v16bf bf;
    uint4 q[2];
    unsigned short us[16];
};

// f32 -> bf16 round-to-nearest-even in the integer domain.
static __device__ __forceinline__ unsigned short f2bf(float f) {
    unsigned int u = __float_as_uint(f);
    u = (u + 0x7FFFu + ((u >> 16) & 1u)) >> 16;
    return (unsigned short)u;
}

static __device__ __forceinline__ v16bf load16bf(const unsigned short* p) {
    Frag f;
    f.q[0] = *reinterpret_cast<const uint4*>(p);
    f.q[1] = *reinterpret_cast<const uint4*>(p + 8);
    return f.bf;
}
// A-fragment style: two 8-half chunks 16 elements apart in K
static __device__ __forceinline__ v16bf load8p8bf(const unsigned short* p0,
                                                  const unsigned short* p1) {
    Frag f;
    f.q[0] = *reinterpret_cast<const uint4*>(p0);
    f.q[1] = *reinterpret_cast<const uint4*>(p1);
    return f.bf;
}

#if defined(__HIP_DEVICE_COMPILE__) && __has_builtin(__builtin_amdgcn_tensor_load_to_lds)
#define USE_TDM 1
#else
#define USE_TDM 0
#endif

#if USE_TDM
// Issue a 2D TDM tile load: tile (tile_d0 x tile_d1) of a row-major tensor
// whose innermost dim has length tensor_d0 and row stride stride0 (elements,
// 2 bytes each).  gptr points at the tile start.  D# packing per ISA 8.3/8.4.
static __device__ __forceinline__ void tdm_load_2d(
    unsigned lds_off, const void* gptr,
    unsigned tensor_d0, unsigned tensor_d1,
    unsigned tile_d0, unsigned tile_d1,
    unsigned long long stride0)
{
    unsigned long long ga = (unsigned long long)(uintptr_t)gptr;
    u32x4 g0;
    g0[0] = 1u;                                            // count=1 (valid)
    g0[1] = lds_off;                                       // lds_addr
    g0[2] = (unsigned)(ga & 0xFFFFFFFFu);                  // global_addr lo
    g0[3] = (unsigned)((ga >> 32) & 0x01FFFFFFu) | 0x80000000u; // hi + type=2
    i32x8 g1;
    g1[0] = (int)(1u << 16);                               // data_size=1 (2B)
    g1[1] = (int)((tensor_d0 & 0xFFFFu) << 16);            // tensor_dim0 lo16
    g1[2] = (int)(((tensor_d0 >> 16) & 0xFFFFu) |
                  ((tensor_d1 & 0xFFFFu) << 16));          // d0 hi / d1 lo
    g1[3] = (int)(((tensor_d1 >> 16) & 0xFFFFu) |
                  ((tile_d0 & 0xFFFFu) << 16));            // d1 hi / tile_d0
    g1[4] = (int)(tile_d1 & 0xFFFFu);                      // tile_d1, tile_d2=0
    g1[5] = (int)(unsigned)(stride0 & 0xFFFFFFFFull);      // dim0 stride lo32
    g1[6] = (int)(unsigned)((stride0 >> 32) & 0xFFFFull);  // stride hi16
    g1[7] = 0;
    i32x4 z4 = {0, 0, 0, 0};
#if __clang_major__ >= 23
    i32x8 z8 = {0, 0, 0, 0, 0, 0, 0, 0};
    __builtin_amdgcn_tensor_load_to_lds(g0, g1, z4, z4, z8, 0);
#else
    __builtin_amdgcn_tensor_load_to_lds(g0, g1, z4, z4, 0);
#endif
}
#endif

// ---------------------------------------------------------------------------
// Kernel 1: per-head projections  X[b,l,h,:] @ W[h]^T  for Q, K, V.
// One wave = one 16-row x 64-col output tile of one tensor.
// Q,K out: [B,H,L,D] bf16 (row-major).  V out: [B,H,D,L] bf16 (transposed).
// ---------------------------------------------------------------------------
__global__ __launch_bounds__(256) void proj_kernel(
    const float* __restrict__ vals, const float* __restrict__ keys,
    const float* __restrict__ qry,
    const float* __restrict__ Wv, const float* __restrict__ Wk,
    const float* __restrict__ Wq,
    unsigned short* __restrict__ Qp, unsigned short* __restrict__ Kp,
    unsigned short* __restrict__ Vp)
{
    const int wave = (blockIdx.x * blockDim.x + threadIdx.x) >> 5;
    const int lane = threadIdx.x & 31;
    const int nLt  = Lx / 16;

    int ltile = wave % nLt;
    int tmp   = wave / nLt;
    int h     = tmp % Hx;  tmp /= Hx;
    int b     = tmp % Bx;
    int t     = tmp / Bx;                         // 0=Q 1=K 2=V

    const float* X = (t == 0) ? qry : (t == 1) ? keys : vals;
    const float* W = (t == 0) ? Wq  : (t == 1) ? Wk  : Wv;

    const int mrow = lane & 15;
    const int half = lane >> 4;
    const int kbA  = half * 16;
    const int kbB  = half * 16;

    const int l = ltile * 16 + mrow;
    const float* xrow = X + ((size_t)(b * Lx + l) * Ex) + h * Dx;
    Frag a[2];
#pragma unroll
    for (int s = 0; s < 2; ++s) {
#pragma unroll
        for (int j = 0; j < 8; ++j) {
            a[s].us[j]     = f2bf(xrow[s * 32 + kbA + j]);
            a[s].us[j + 8] = f2bf(xrow[s * 32 + kbA + 8 + j]);
        }
    }

    const float* Wh   = W + (size_t)h * Dx * Dx;
    const int    ncol = lane & 15;

#pragma unroll
    for (int n = 0; n < 4; ++n) {
        const int e = n * 16 + ncol;
        const float* wrow = Wh + (size_t)e * Dx + kbB;
        Frag bb[2];
#pragma unroll
        for (int s = 0; s < 2; ++s)
#pragma unroll
            for (int j = 0; j < 16; ++j)
                bb[s].us[j] = f2bf(wrow[s * 32 + j]);

        v8f c = {};
        c = __builtin_amdgcn_wmma_f32_16x16x32_bf16(false, a[0].bf, false,
                                                    bb[0].bf, (short)0, c,
                                                    false, false);
        c = __builtin_amdgcn_wmma_f32_16x16x32_bf16(false, a[1].bf, false,
                                                    bb[1].bf, (short)0, c,
                                                    false, false);
#pragma unroll
        for (int r = 0; r < 8; ++r) {
            const int lr = ltile * 16 + r + half * 8;
            const int d  = n * 16 + ncol;
            const unsigned short v = f2bf(c[r]);
            if (t == 0)
                Qp[(((size_t)(b * Hx + h) * Lx + lr) * Dx) + d] = v;
            else if (t == 1)
                Kp[(((size_t)(b * Hx + h) * Lx + lr) * Dx) + d] = v;
            else
                Vp[(((size_t)(b * Hx + h) * Dx + d) * Lx) + lr] = v;   // V^T
        }
    }
}

// ---------------------------------------------------------------------------
// Kernel 2: flash attention.  Block = 4 waves = 64 Q rows of one (b,h).
// K/V tiles (32 keys) streamed into LDS by the TDM, double-buffered.
// ---------------------------------------------------------------------------
__global__ __launch_bounds__(128) void flash_kernel(
    const unsigned short* __restrict__ Qp, const unsigned short* __restrict__ Kp,
    const unsigned short* __restrict__ Vp, const int* __restrict__ mask,
    unsigned short* __restrict__ Ao)
{
    __shared__ unsigned short ktile[2][32 * Dx];  // [key][d]   pitch 64, 4KB
    __shared__ unsigned short vtile[2][Dx * 32];  // [d][key]   pitch 32, 4KB
    __shared__ unsigned short pbuf[4][16 * 40];   // P transpose, pitch 40

    const int wv   = threadIdx.x >> 5;
    const int lane = threadIdx.x & 31;
    const int half = lane >> 4;
    const int col  = lane & 15;

    const int nQb  = Lx / 64;
    const int bh   = blockIdx.x / nQb;
    const int qblk = blockIdx.x % nQb;
    const int b    = bh / Hx;
    const int h    = bh % Hx;
    const int qbase = qblk * 64 + wv * 16;

    const float scale = 0.125f;                   // 1/sqrt(64)

    const unsigned short* kbase0 = Kp + (size_t)(b * Hx + h) * Lx * Dx;
    const unsigned short* vbase0 = Vp + (size_t)(b * Hx + h) * Dx * Lx;
    const int* mb = mask + (size_t)b * Lx * Lx;   // [B,1,L,L]

    // ---- load Q A-fragments (16 rows x K=64) ----
    const unsigned short* qrow =
        Qp + ((size_t)(b * Hx + h) * Lx + qbase + col) * Dx;
    const int kbA = half * 16;
    v16bf qa0 = load16bf(qrow + kbA);
    v16bf qa1 = load16bf(qrow + 32 + kbA);

    v8f o[4] = {};
    float mrun[8], lrun[8];
#pragma unroll
    for (int r = 0; r < 8; ++r) { mrun[r] = -3.0e38f; lrun[r] = 0.0f; }

    const int NT = Lx / 32;

#if USE_TDM
    // Wave 0 drives the Tensor Data Mover: tile pair (K, V^T) per key tile.
    auto issue = [&](int buf, int kt) {
        tdm_load_2d((unsigned)(uintptr_t)&ktile[buf][0],
                    kbase0 + (size_t)(kt * 32) * Dx,
                    Dx, Lx, Dx, 32, (unsigned long long)Dx);
        tdm_load_2d((unsigned)(uintptr_t)&vtile[buf][0],
                    vbase0 + kt * 32,
                    Lx, Dx, 32, Dx, (unsigned long long)Lx);
    };
    if (wv == 0) issue(0, 0);
#endif

    for (int kt = 0; kt < NT; ++kt) {
        const int k0 = kt * 32;
#if USE_TDM
        const int cur = kt & 1;
        if (wv == 0) {
            if (kt + 1 < NT) {
                issue(1 - cur, kt + 1);
                __builtin_amdgcn_s_wait_tensorcnt(2);  // current pair retired
            } else {
                __builtin_amdgcn_s_wait_tensorcnt(0);
            }
        }
        __syncthreads();
#else
        const int cur = 0;
        // Cooperative fallback: 128 threads copy both 4KB tiles.
        {
            const int tid = threadIdx.x;
            for (int i = tid; i < 32 * Dx / 8; i += 128) {
                reinterpret_cast<uint4*>(&ktile[0][0])[i] =
                    reinterpret_cast<const uint4*>(kbase0 +
                        (size_t)k0 * Dx)[i];
            }
            for (int i = tid; i < Dx * 32 / 8; i += 128) {
                const int dr = i / 4;          // d row, 4 uint4 per 32 keys? no:
                const int cw = i % 4;          // 32 halves = 4 uint4 chunks
                reinterpret_cast<uint4*>(&vtile[0][dr * 32])[cw] =
                    reinterpret_cast<const uint4*>(vbase0 +
                        (size_t)dr * Lx + k0)[cw];
            }
            __syncthreads();
        }
#endif

        // keep a global_prefetch_b8 stream on the mask rows
        if (kt + 1 < NT)
            __builtin_prefetch(mb + (size_t)(qbase + col) * Lx + k0 + 32, 0, 0);

        // ---- S = Q @ K^T from LDS K tile ----
        v8f s0 = {}, s1 = {};
        {
            const unsigned short* kl = &ktile[cur][0];
            const unsigned short* kr0 = kl + (size_t)col * Dx + half * 16;
            const unsigned short* kr1 = kl + (size_t)(16 + col) * Dx + half * 16;
            v16bf b00 = load16bf(kr0);
            v16bf b01 = load16bf(kr0 + 32);
            v16bf b10 = load16bf(kr1);
            v16bf b11 = load16bf(kr1 + 32);
            s0 = __builtin_amdgcn_wmma_f32_16x16x32_bf16(false, qa0, false, b00,
                                                         (short)0, s0, false, false);
            s0 = __builtin_amdgcn_wmma_f32_16x16x32_bf16(false, qa1, false, b01,
                                                         (short)0, s0, false, false);
            s1 = __builtin_amdgcn_wmma_f32_16x16x32_bf16(false, qa0, false, b10,
                                                         (short)0, s1, false, false);
            s1 = __builtin_amdgcn_wmma_f32_16x16x32_bf16(false, qa1, false, b11,
                                                         (short)0, s1, false, false);
        }

        // ---- mask + scale + online softmax update ----
#pragma unroll
        for (int r = 0; r < 8; ++r) {
            const int q = qbase + r + half * 8;
            const int m0 = mb[(size_t)q * Lx + k0 + col];
            const int m1 = mb[(size_t)q * Lx + k0 + 16 + col];
            float e0 = ((m0 == 0) ? -1.0e20f : s0[r]) * scale;
            float e1 = ((m1 == 0) ? -1.0e20f : s1[r]) * scale;

            float tm = fmaxf(e0, e1);
#pragma unroll
            for (int off = 8; off >= 1; off >>= 1)
                tm = fmaxf(tm, __shfl_xor(tm, off, 16));
            const float mnew  = fmaxf(mrun[r], tm);
            const float alpha = __expf(mrun[r] - mnew);
            const float p0 = __expf(e0 - mnew);
            const float p1 = __expf(e1 - mnew);
            float rs = p0 + p1;
#pragma unroll
            for (int off = 8; off >= 1; off >>= 1)
                rs += __shfl_xor(rs, off, 16);
            lrun[r] = lrun[r] * alpha + rs;
            mrun[r] = mnew;
#pragma unroll
            for (int nt = 0; nt < 4; ++nt) o[nt][r] *= alpha;

            const int prow = r + half * 8;
            pbuf[wv][prow * 40 + col]      = f2bf(p0);
            pbuf[wv][prow * 40 + col + 16] = f2bf(p1);
        }

        asm volatile("s_wait_dscnt 0" ::: "memory");

        // ---- P as A-fragment (row = lane&15, K = 32 keys) ----
        const unsigned short* pr = &pbuf[wv][col * 40 + kbA];
        v16bf pa = load16bf(pr);

        // ---- O += P @ V from LDS V^T tile ----
#pragma unroll
        for (int nt = 0; nt < 4; ++nt) {
            const unsigned short* vp =
                &vtile[cur][(size_t)(nt * 16 + col) * 32 + half * 16];
            v16bf vb = load16bf(vp);
            o[nt] = __builtin_amdgcn_wmma_f32_16x16x32_bf16(false, pa, false, vb,
                                                            (short)0, o[nt],
                                                            false, false);
        }

        __syncthreads();   // tiles consumed; next TDM issue may overwrite
    }

    // ---- normalize + store Ao[B,L,E] (E = h*D + d) as bf16 ----
#pragma unroll
    for (int r = 0; r < 8; ++r) {
        const float inv = 1.0f / lrun[r];
        const int q = qbase + r + half * 8;
#pragma unroll
        for (int nt = 0; nt < 4; ++nt) {
            const int d = nt * 16 + col;
            Ao[((size_t)(b * Lx + q) * Hx + h) * Dx + d] = f2bf(o[nt][r] * inv);
        }
    }
}

// ---------------------------------------------------------------------------
// Kernel 2.5: one-time Wo f32 -> bf16 conversion (keeps it out of the GEMM
// inner loop).
// ---------------------------------------------------------------------------
__global__ __launch_bounds__(256) void cvt_wo_kernel(
    const float* __restrict__ Wo, unsigned short* __restrict__ Wob)
{
    const int i = (blockIdx.x * blockDim.x + threadIdx.x) * 4;
    const float4 f = *reinterpret_cast<const float4*>(Wo + i);
    uint2 o;
    o.x = (unsigned)f2bf(f.x) | ((unsigned)f2bf(f.y) << 16);
    o.y = (unsigned)f2bf(f.z) | ((unsigned)f2bf(f.w) << 16);
    *reinterpret_cast<uint2*>(Wob + i) = o;
}

// ---------------------------------------------------------------------------
// Kernel 3: out = Ao @ Wo^T + bo, f32 output.  One wave = 16 x 64 tile.
// Inner loop: 2 b128 loads + 1 WMMA (no conversions).
// ---------------------------------------------------------------------------
__global__ __launch_bounds__(256) void outproj_kernel(
    const unsigned short* __restrict__ Ao, const unsigned short* __restrict__ Wob,
    const float* __restrict__ bo, float* __restrict__ out)
{
    const int wave = (blockIdx.x * blockDim.x + threadIdx.x) >> 5;
    const int lane = threadIdx.x & 31;
    const int half = lane >> 4;
    const int col  = lane & 15;
    const int kbA  = half * 16;
    const int kbB  = half * 16;

    const int nNg = Ex / 64;
    const int mt  = wave / nNg;
    const int ng  = wave % nNg;

    const int row = mt * 16 + col;
    const unsigned short* arow = Ao + (size_t)row * Ex;

    v8f c[4] = {};
    for (int s = 0; s < Ex / 32; ++s) {
        const unsigned short* ap = arow + s * 32 + kbA;
        v16bf a = load16bf(ap);
#pragma unroll
        for (int n = 0; n < 4; ++n) {
            const int e = ng * 64 + n * 16 + col;
            v16bf bb = load16bf(Wob + (size_t)e * Ex + s * 32 + kbB);
            c[n] = __builtin_amdgcn_wmma_f32_16x16x32_bf16(false, a, false, bb,
                                                           (short)0, c[n],
                                                           false, false);
        }
    }

#pragma unroll
    for (int n = 0; n < 4; ++n) {
        const int e = ng * 64 + n * 16 + col;
        const float bias = bo[e];
#pragma unroll
        for (int r = 0; r < 8; ++r) {
            const int rr = mt * 16 + r + half * 8;
            out[(size_t)rr * Ex + e] = c[n][r] + bias;
        }
    }
}

// ---------------------------------------------------------------------------
extern "C" void kernel_launch(void* const* d_in, const int* in_sizes, int n_in,
                              void* d_out, int out_size, void* d_ws,
                              size_t ws_size, hipStream_t stream)
{
    const float* vals = (const float*)d_in[0];
    const float* keys = (const float*)d_in[1];
    const float* qry  = (const float*)d_in[2];
    const int*   mask = (const int*)d_in[3];
    // d_in[4] = size (compile-time constant Bx)
    const float* Wv = (const float*)d_in[5];
    const float* Wk = (const float*)d_in[6];
    const float* Wq = (const float*)d_in[7];
    const float* Wo = (const float*)d_in[8];
    const float* bo = (const float*)d_in[9];
    float* out = (float*)d_out;

    // Workspace: 4 x 16MB bf16 buffers + 2MB bf16 Wo  (66 MB total)
    const size_t N1 = (size_t)Bx * Hx * Lx * Dx;          // 8M halves
    unsigned short* Qp  = (unsigned short*)d_ws;
    unsigned short* Kp  = Qp + N1;
    unsigned short* Vp  = Kp + N1;                        // [B,H,D,L]
    unsigned short* Ao  = Vp + N1;                        // [B,L,E]
    unsigned short* Wob = Ao + N1;                        // [E,E]

    // 0) Wo -> bf16 once
    cvt_wo_kernel<<<(Ex * Ex / 4) / 256, 256, 0, stream>>>(Wo, Wob);

    // 1) projections: 3 tensors * B*H*(L/16) waves = 24576 waves
    {
        const int waves = 3 * Bx * Hx * (Lx / 16);
        const int blocks = waves * 32 / 256;
        proj_kernel<<<blocks, 256, 0, stream>>>(vals, keys, qry, Wv, Wk, Wq,
                                                Qp, Kp, Vp);
    }
    // 2) flash attention: B*H*(L/64) blocks of 4 waves
    {
        const int blocks = Bx * Hx * (Lx / 64);
        flash_kernel<<<blocks, 128, 0, stream>>>(Qp, Kp, Vp, mask, Ao);
    }
    // 3) output projection: (B*L/16)*(E/64) waves = 8192 waves
    {
        const int waves = (Bx * Lx / 16) * (Ex / 64);
        const int blocks = waves * 32 / 256;
        outproj_kernel<<<blocks, 256, 0, stream>>>(Ao, Wob, bo, out);
    }
}
